// PRTAttention_8280696946856
// MI455X (gfx1250) — compile-verified, matrix operand order checked by reading, not run on hardware
//
#include <hip/hip_runtime.h>
#include <hip/hip_bf16.h>

typedef __bf16 bf16;
typedef __attribute__((ext_vector_type(16))) __bf16 bf16x16;
typedef __attribute__((ext_vector_type(8)))  __bf16 bf16x8;
typedef __attribute__((ext_vector_type(8)))  float  f32x8;

#define DEV static __device__ __forceinline__

constexpr int BATCH  = 16;
constexpr int GRID_S = 32;
constexpr int NTOK   = GRID_S * GRID_S + 1;   // 1025
constexpr int HEADS  = 6;
constexpr int HDIM   = 64;
constexpr int CDIM   = HEADS * HDIM;          // 384
constexpr int C3     = 3 * CDIM;              // 1152
constexpr int MROWS  = BATCH * NTOK;          // 16400 (divisible by 16)
constexpr int RANGE  = 2;
constexpr float SCALE = 0.125f;               // 64^-0.5
constexpr int NT_Q   = (NTOK + 15) / 16;      // 65 query tiles
constexpr int NPAD   = 17 * 64;               // 1088: padded key dim of V^T rows

// ---------- helpers ----------
DEV bf16 f2bf(float f) {
    union { float f; unsigned u; } c{f};
    unsigned r = c.u + 0x7FFFu + ((c.u >> 16) & 1u);   // round-to-nearest-even
    union { unsigned short s; bf16 b; } o{(unsigned short)(r >> 16)};
    return o.b;
}

DEV bf16x16 make16(bf16x8 lo, bf16x8 hi) {
    union { bf16x16 v; bf16x8 h[2]; } u;
    u.h[0] = lo; u.h[1] = hi;
    return u.v;
}

// 16-element bf16 fragment: 8 contiguous at base, 8 at base+16.
// ISA 16-bit A/B layout: lane's row/col with K={0..7,16..23} (lanes 0-15)
// or {8..15,24..31} (lanes 16-31) when `base` already includes half*8.
DEV bf16x16 load_frag(const bf16* base) {
    bf16x8 lo = *(const bf16x8*)base;
    bf16x8 hi = *(const bf16x8*)(base + 16);
    return make16(lo, hi);
}

DEV f32x8 wmma_bf16(bf16x16 a, bf16x16 b, f32x8 c) {
    return __builtin_amdgcn_wmma_f32_16x16x32_bf16(
        false, a, false, b, (short)0, c, false, false);
}

DEV float mask_add(int q, int m) {
    if (q >= NTOK || m >= NTOK) return -1e30f;
    if (q == 0 || m == 0) return 0.0f;
    int pq = q - 1, pm = m - 1;
    int dr = (pq >> 5) - (pm >> 5); dr = dr < 0 ? -dr : dr;
    int dc = (pq & 31) - (pm & 31); dc = dc < 0 ? -dc : dc;
    int ch = dr > dc ? dr : dc;
    return ch <= RANGE ? 0.0f : -1e30f;
}

// ---------- kernel 1: fp32 -> bf16 conversion ----------
__global__ void __launch_bounds__(256)
convert_kernel(const float* __restrict__ x, const float* __restrict__ wq,
               const float* __restrict__ wp, bf16* __restrict__ xb,
               bf16* __restrict__ wqb, bf16* __restrict__ wpb) {
    const int stride = gridDim.x * blockDim.x;
    const int tid = blockIdx.x * blockDim.x + threadIdx.x;
    for (int i = tid; i < MROWS * CDIM; i += stride) xb[i]  = f2bf(x[i]);
    for (int i = tid; i < C3 * CDIM;    i += stride) wqb[i] = f2bf(wq[i]);
    for (int i = tid; i < CDIM * CDIM;  i += stride) wpb[i] = f2bf(wp[i]);
}

// ---------- kernel 2: QKV GEMM  Y = X @ W^T + b, scatter to Q/K/V ----------
__global__ void __launch_bounds__(256)
qkv_gemm_kernel(const bf16* __restrict__ X, const bf16* __restrict__ W,
                const float* __restrict__ bias, bf16* __restrict__ Q,
                bf16* __restrict__ K, bf16* __restrict__ V) {
    const int wave = threadIdx.x >> 5;
    const int lane = threadIdx.x & 31;
    const int half = lane >> 4;
    const int l    = lane & 15;
    constexpr int NTILES = C3 / 16;                 // 72
    const int w = blockIdx.x * 8 + wave;
    if (w >= (MROWS / 16) * NTILES) return;         // wave-uniform
    const int mt = w / NTILES, nt = w % NTILES;
    const int m0 = mt * 16,    n0 = nt * 16;

    const bf16* arow = X + (size_t)(m0 + l) * CDIM + half * 8;
    const bf16* brow = W + (size_t)(n0 + l) * CDIM + half * 8;

    f32x8 acc;
    #pragma unroll
    for (int r = 0; r < 8; ++r) acc[r] = 0.0f;

    #pragma unroll
    for (int kk = 0; kk < CDIM; kk += 32) {
        if (kk + 32 < CDIM) __builtin_prefetch(arow + kk + 32, 0, 3);
        acc = wmma_bf16(load_frag(arow + kk), load_frag(brow + kk), acc);
    }

    // column j -> (s, h, d) per reshape(B,N,3,H,D)
    const int j = n0 + l;
    const int s = j / CDIM;
    const int rem = j - s * CDIM;
    const int h = rem >> 6, d = rem & 63;
    bf16* dst = (s == 0) ? Q : (s == 1) ? K : V;
    const float bj = bias[j];
    #pragma unroll
    for (int r = 0; r < 8; ++r) {
        const int row = m0 + (half ? r + 8 : r);          // global row in [0,16400)
        const int b_ = row / NTOK;
        const int n_ = row - b_ * NTOK;
        dst[(((size_t)b_ * HEADS + h) * NTOK + n_) * HDIM + d] = f2bf(acc[r] + bj);
    }
}

// ---------- kernel 2b: V -> V^T (per head), LDS-tiled, coalesced both ways ----
// Vt[bh][d][k] with rows padded to NPAD=1088 keys (17 tiles of 64).
__global__ void __launch_bounds__(256)
vtrans_kernel(const bf16* __restrict__ V, bf16* __restrict__ Vt) {
    __shared__ __align__(16) bf16 tile[64 * 80];          // 64 keys x 64 dims (+pad)
    const int bh = blockIdx.x / 17;
    const int kt = blockIdx.x % 17;
    const int n0 = kt * 64;
    const int t  = threadIdx.x;

    {   // load 64 key rows x 64 dims, 16B per thread per row-half
        const int rbase = t >> 3;         // 0..31
        const int c8    = (t & 7) * 8;    // 0,8,...,56
        #pragma unroll
        for (int i = 0; i < 2; ++i) {
            const int nl = rbase + i * 32;
            int n = n0 + nl; if (n > NTOK - 1) n = NTOK - 1;   // clamp; masked later
            *(bf16x8*)(tile + nl * 80 + c8) =
                *(const bf16x8*)(V + ((size_t)bh * NTOK + n) * HDIM + c8);
        }
    }
    __syncthreads();
    {   // write 64 dim rows x 64 keys, 32B contiguous per thread
        const int d  = t >> 2;            // 0..63
        const int k0 = (t & 3) * 16;      // 0,16,32,48
        union { bf16x8 v[2]; bf16 e[16]; } row;
        #pragma unroll
        for (int j = 0; j < 16; ++j) row.e[j] = tile[(k0 + j) * 80 + d];
        bf16* dst = Vt + ((size_t)bh * HDIM + d) * NPAD + n0 + k0;
        *(bf16x8*)(dst)     = row.v[0];
        *(bf16x8*)(dst + 8) = row.v[1];
    }
}

// ---------- kernel 3: block-sparse flash attention ----------
// one wave = one (b, h, 16-query tile); key chunks of 32, masked chunks skipped
__global__ void __launch_bounds__(128)
attn_kernel(const bf16* __restrict__ Qb, const bf16* __restrict__ Kb,
            const bf16* __restrict__ Vt, bf16* __restrict__ att) {
    __shared__ __align__(16) bf16 lds[4 * 16 * 32];       // per-wave P transpose tile
    const int wave = threadIdx.x >> 5;
    const int lane = threadIdx.x & 31;
    const int half = lane >> 4;
    const int l    = lane & 15;
    bf16* ptile = lds + wave * (16 * 32);

    const int task = blockIdx.x * 4 + wave;
    if (task >= BATCH * HEADS * NT_Q) return;             // wave-uniform
    const int qt = task % NT_Q;
    const int bh = task / NT_Q;                           // b*HEADS + h
    const int q0 = qt * 16;

    const bf16* Qbase  = Qb + (size_t)bh * NTOK * HDIM;
    const bf16* Kbase  = Kb + (size_t)bh * NTOK * HDIM;
    const bf16* VtBase = Vt + (size_t)bh * HDIM * NPAD;

    // Q fragments: 16x64 = two K=32 A-fragments, resident for whole tile
    int qr = q0 + l; if (qr > NTOK - 1) qr = NTOK - 1;
    const bf16* qp = Qbase + (size_t)qr * HDIM + half * 8;
    const bf16x16 qa0 = load_frag(qp);
    const bf16x16 qa1 = load_frag(qp + 32);

    float mrow[8], lrow[8];
    f32x8 oacc[4];
    #pragma unroll
    for (int r = 0; r < 8; ++r) { mrow[r] = -1e30f; lrow[r] = 0.0f; }
    #pragma unroll
    for (int dg = 0; dg < 4; ++dg)
        #pragma unroll
        for (int r = 0; r < 8; ++r) oacc[dg][r] = 0.0f;

    // reachable key-chunk range from the Chebyshev mask
    int rmin = 0, rmax = 0;
    if (qt != 0) {
        const int pmin = q0 - 1;
        int pm = (q0 + 15 < NTOK - 1 ? q0 + 15 : NTOK - 1) - 1;
        rmin = (pmin >> 5) - RANGE; if (rmin < 0) rmin = 0;
        rmax = (pm   >> 5) + RANGE; if (rmax > GRID_S - 1) rmax = GRID_S - 1;
    }

    for (int phase = 0; phase < 2; ++phase) {
        int clo, chi;
        if (qt == 0) { if (phase) break; clo = 0; chi = 32; }      // CLS row: dense
        else if (phase == 0) { clo = 0; chi = 0; }                 // CLS key chunk
        else { clo = (rmin > 1 ? rmin : 1); chi = rmax + 1; if (clo > chi) break; }

        for (int c = clo; c <= chi; ++c) {
            const int kbase = c * 32;

            // S = Q @ K^T over 32 keys -> two 16x16 C-fragments
            f32x8 sc0, sc1;
            #pragma unroll
            for (int r = 0; r < 8; ++r) { sc0[r] = 0.0f; sc1[r] = 0.0f; }
            {
                int mg0 = kbase + l, mg1 = kbase + 16 + l;
                if (mg0 > NTOK - 1) mg0 = NTOK - 1;
                if (mg1 > NTOK - 1) mg1 = NTOK - 1;
                const bf16* kp0 = Kbase + (size_t)mg0 * HDIM + half * 8;
                const bf16* kp1 = Kbase + (size_t)mg1 * HDIM + half * 8;
                sc0 = wmma_bf16(qa0, load_frag(kp0),      sc0);
                sc0 = wmma_bf16(qa1, load_frag(kp0 + 32), sc0);
                sc1 = wmma_bf16(qa0, load_frag(kp1),      sc1);
                sc1 = wmma_bf16(qa1, load_frag(kp1 + 32), sc1);
            }

            // V B-fragments straight from V^T: contiguous 16B pairs per lane
            bf16x16 vfrag[4];
            #pragma unroll
            for (int dg = 0; dg < 4; ++dg) {
                const bf16* vp = VtBase + (size_t)(16 * dg + l) * NPAD
                                 + kbase + half * 8;
                vfrag[dg] = load_frag(vp);
            }

            // mask + scale + online softmax (rows live in 16-lane groups)
            float pv0[8], pv1[8];
            #pragma unroll
            for (int r = 0; r < 8; ++r) {
                const int q = q0 + (half ? r + 8 : r);
                float s0 = sc0[r] * SCALE + mask_add(q, kbase + l);
                float s1 = sc1[r] * SCALE + mask_add(q, kbase + 16 + l);
                float rm = fmaxf(s0, s1);
                rm = fmaxf(rm, __shfl_xor(rm, 1, 32));
                rm = fmaxf(rm, __shfl_xor(rm, 2, 32));
                rm = fmaxf(rm, __shfl_xor(rm, 4, 32));
                rm = fmaxf(rm, __shfl_xor(rm, 8, 32));
                const float mnew  = fmaxf(mrow[r], rm);
                const float alpha = __expf(mrow[r] - mnew);
                const float p0 = __expf(s0 - mnew);
                const float p1 = __expf(s1 - mnew);
                float rs = p0 + p1;
                rs += __shfl_xor(rs, 1, 32);
                rs += __shfl_xor(rs, 2, 32);
                rs += __shfl_xor(rs, 4, 32);
                rs += __shfl_xor(rs, 8, 32);
                lrow[r] = lrow[r] * alpha + rs;
                mrow[r] = mnew;
                #pragma unroll
                for (int dg = 0; dg < 4; ++dg) oacc[dg][r] *= alpha;
                pv0[r] = p0; pv1[r] = p1;
            }

            // transpose P (C-layout -> A-layout) through LDS
            #pragma unroll
            for (int r = 0; r < 8; ++r) {
                const int row = half ? r + 8 : r;
                ptile[row * 32 + l]      = f2bf(pv0[r]);
                ptile[row * 32 + 16 + l] = f2bf(pv1[r]);
            }
            const bf16* pr = ptile + l * 32 + half * 8;
            const bf16x16 pa = make16(*(const bf16x8*)pr, *(const bf16x8*)(pr + 16));

            // O += P @ V
            #pragma unroll
            for (int dg = 0; dg < 4; ++dg)
                oacc[dg] = wmma_bf16(pa, vfrag[dg], oacc[dg]);
        }
    }

    // normalize + store to [B, N, C] bf16 (C index = h*64 + d)
    const int b_ = bh / HEADS, h_ = bh % HEADS;
    #pragma unroll
    for (int dg = 0; dg < 4; ++dg) {
        #pragma unroll
        for (int r = 0; r < 8; ++r) {
            const int q = q0 + (half ? r + 8 : r);
            if (q < NTOK) {
                const float denom = lrow[r];
                const float val = denom > 0.0f ? oacc[dg][r] / denom : 0.0f;
                att[((size_t)b_ * NTOK + q) * CDIM + h_ * HDIM + 16 * dg + l] = f2bf(val);
            }
        }
    }
}

// ---------- kernel 4: output projection  out = A @ Wp^T + b ----------
__global__ void __launch_bounds__(256)
proj_gemm_kernel(const bf16* __restrict__ A, const bf16* __restrict__ W,
                 const float* __restrict__ bias, float* __restrict__ out) {
    const int wave = threadIdx.x >> 5;
    const int lane = threadIdx.x & 31;
    const int half = lane >> 4;
    const int l    = lane & 15;
    constexpr int NTILES = CDIM / 16;                // 24
    const int w = blockIdx.x * 8 + wave;
    if (w >= (MROWS / 16) * NTILES) return;
    const int mt = w / NTILES, nt = w % NTILES;
    const int m0 = mt * 16,    n0 = nt * 16;

    const bf16* arow = A + (size_t)(m0 + l) * CDIM + half * 8;
    const bf16* brow = W + (size_t)(n0 + l) * CDIM + half * 8;

    f32x8 acc;
    #pragma unroll
    for (int r = 0; r < 8; ++r) acc[r] = 0.0f;

    #pragma unroll
    for (int kk = 0; kk < CDIM; kk += 32) {
        if (kk + 32 < CDIM) __builtin_prefetch(arow + kk + 32, 0, 3);
        acc = wmma_bf16(load_frag(arow + kk), load_frag(brow + kk), acc);
    }

    const int j = n0 + l;
    const float bj = bias[j];
    #pragma unroll
    for (int r = 0; r < 8; ++r) {
        const int row = m0 + (half ? r + 8 : r);
        out[(size_t)row * CDIM + j] = acc[r] + bj;
    }
}

// ---------- host launcher ----------
extern "C" void kernel_launch(void* const* d_in, const int* in_sizes, int n_in,
                              void* d_out, int out_size, void* d_ws, size_t ws_size,
                              hipStream_t stream) {
    const float* x      = (const float*)d_in[0];
    const float* qkv_w  = (const float*)d_in[1];
    const float* qkv_b  = (const float*)d_in[2];
    const float* proj_w = (const float*)d_in[3];
    const float* proj_b = (const float*)d_in[4];
    float* out = (float*)d_out;

    char* ws = (char*)d_ws;
    size_t off = 0;
    bf16* xb   = (bf16*)(ws + off); off += (size_t)MROWS * CDIM * sizeof(bf16);
    bf16* wqb  = (bf16*)(ws + off); off += (size_t)C3 * CDIM * sizeof(bf16);
    bf16* wpb  = (bf16*)(ws + off); off += (size_t)CDIM * CDIM * sizeof(bf16);
    bf16* Qbuf = (bf16*)(ws + off); off += (size_t)BATCH * HEADS * NTOK * HDIM * sizeof(bf16);
    bf16* Kbuf = (bf16*)(ws + off); off += (size_t)BATCH * HEADS * NTOK * HDIM * sizeof(bf16);
    bf16* Vbuf = (bf16*)(ws + off); off += (size_t)BATCH * HEADS * NTOK * HDIM * sizeof(bf16);
    bf16* VtBf = (bf16*)(ws + off); off += (size_t)BATCH * HEADS * HDIM * NPAD * sizeof(bf16);
    bf16* attb = (bf16*)(ws + off); off += (size_t)MROWS * CDIM * sizeof(bf16);

    convert_kernel<<<1024, 256, 0, stream>>>(x, qkv_w, proj_w, xb, wqb, wpb);

    {   // 1025 x 72 wave tiles, 8 waves / block
        const int waves = (MROWS / 16) * (C3 / 16);
        qkv_gemm_kernel<<<(waves + 7) / 8, 256, 0, stream>>>(
            xb, wqb, qkv_b, Qbuf, Kbuf, Vbuf);
    }
    // per-head V transpose: 96 heads x 17 key tiles
    vtrans_kernel<<<BATCH * HEADS * 17, 256, 0, stream>>>(Vbuf, VtBf);
    {   // 16 * 6 * 65 wave tasks, 4 waves / block
        const int tasks = BATCH * HEADS * NT_Q;
        attn_kernel<<<(tasks + 3) / 4, 128, 0, stream>>>(Qbuf, Kbuf, VtBf, attb);
    }
    {   // 1025 x 24 wave tiles
        const int waves = (MROWS / 16) * (CDIM / 16);
        proj_gemm_kernel<<<(waves + 7) / 8, 256, 0, stream>>>(attb, wpb, proj_b, out);
    }
}
